// SlotAttention_model_55972013801575
// MI455X (gfx1250) — compile-verified, matrix operand order checked by reading, not run on hardware
//
#include <hip/hip_runtime.h>

typedef __bf16 bf16;
typedef __attribute__((ext_vector_type(16))) __bf16 v16bf;
typedef __attribute__((ext_vector_type(8)))  float  v8f;

constexpr int B = 32, N = 16384, D = 64, NS = 7, H = 128;
constexpr float SCALE = 0.125f;   // D^-0.5
constexpr float EPS = 1e-8f;

// ---- WMMA bf16 fragment load (ISA 7.12.2 16-bit 16x32 A layout; B mirrored lane<->N) ----
// lane 0-15: row m=lane, halves h=0..7 -> K=0..7, h=8..15 -> K=16..23
// lane 16-31: row m=lane-16, K offset +8
__device__ inline v16bf ldfrag(const bf16* src, int ld) {
  const int lane = threadIdx.x & 31;
  const int m = lane & 15;
  const int kb = (lane >> 4) << 3;
  const bf16* row = src + (size_t)m * ld + kb;
  v16bf f;
#pragma unroll
  for (int h = 0; h < 16; ++h) {
    int k = (h & 7) + ((h & 8) << 1);
    f[h] = row[k];
  }
  return f;
}

#define WMMA_BF16(A, Bm, C) \
  __builtin_amdgcn_wmma_f32_16x16x32_bf16(false, (A), false, (Bm), (short)0, (C), false, false)

// ---------------- K0: slots = mu + sigma * noise ----------------
__global__ void k_init_slots(const float* __restrict__ noise, const float* __restrict__ mu,
                             const float* __restrict__ sg, float* __restrict__ slots) {
  int i = blockIdx.x * 256 + threadIdx.x;
  if (i < B * NS * D) slots[i] = mu[i & 63] + sg[i & 63] * noise[i];
}

// ---------------- K1: LN(inputs) -> k (bf16 row-major), v (bf16 transposed [D][N]) ----------------
__global__ __launch_bounds__(256)
void k_ln_kv(const float* __restrict__ inp, const float* __restrict__ Wk,
             const float* __restrict__ bk, const float* __restrict__ Wv,
             const float* __restrict__ bv, const float* __restrict__ gin,
             const float* __restrict__ bin, bf16* __restrict__ kbuf,
             bf16* __restrict__ vT) {
  __shared__ float xs[128 * 64];
  __shared__ bf16  xn[128 * 64];
  __shared__ bf16  wkl[64 * 64];
  __shared__ bf16  wvl[64 * 64];
  __shared__ float bias[128];
  __shared__ float gg[64], bb[64];

  const int t = threadIdx.x;
  const size_t R0 = (size_t)blockIdx.x * 128;

  for (int i = t; i < 64 * 64; i += 256) { wkl[i] = (bf16)Wk[i]; wvl[i] = (bf16)Wv[i]; }
  if (t < 64) { bias[t] = bk[t]; bias[64 + t] = bv[t]; gg[t] = gin[t]; bb[t] = bin[t]; }
  for (int i = t; i < 128 * 64; i += 256) xs[i] = inp[R0 * 64 + i];
  __syncthreads();

  if (t < 128) {
    const float* r = xs + t * 64;
    float m = 0.f;
#pragma unroll
    for (int d = 0; d < 64; ++d) m += r[d];
    m *= (1.f / 64.f);
    float v = 0.f;
#pragma unroll
    for (int d = 0; d < 64; ++d) { float dd = r[d] - m; v += dd * dd; }
    v *= (1.f / 64.f);
    float rs = rsqrtf(v + 1e-5f);
#pragma unroll
    for (int d = 0; d < 64; ++d) xn[t * 64 + d] = (bf16)((r[d] - m) * rs * gg[d] + bb[d]);
  }
  __syncthreads();

  const int w = t >> 5;
  const int lane = t & 31;
  const int n = lane & 15;
  const int mb = (lane >> 4) << 3;
  const int wr = w * 16;
  v16bf a0 = ldfrag(xn + wr * 64, 64);
  v16bf a1 = ldfrag(xn + wr * 64 + 32, 64);
  const int bidx = (int)(R0 / N);
  const int nb0 = (int)(R0 % N) + wr + mb;

#pragma unroll
  for (int nt = 0; nt < 4; ++nt) {
    v16bf bk0 = ldfrag(wkl + nt * 16 * 64, 64);
    v16bf bk1 = ldfrag(wkl + nt * 16 * 64 + 32, 64);
    v8f ck = {};
    ck = WMMA_BF16(a0, bk0, ck);
    ck = WMMA_BF16(a1, bk1, ck);
    v16bf bv0 = ldfrag(wvl + nt * 16 * 64, 64);
    v16bf bv1 = ldfrag(wvl + nt * 16 * 64 + 32, 64);
    v8f cv = {};
    cv = WMMA_BF16(a0, bv0, cv);
    cv = WMMA_BF16(a1, bv1, cv);
    const int col = nt * 16 + n;
    const float bkc = bias[col], bvc = bias[64 + col];
#pragma unroll
    for (int vv = 0; vv < 8; ++vv) {
      size_t grow = R0 + wr + mb + vv;
      kbuf[grow * 64 + col] = (bf16)(ck[vv] + bkc);
      vT[((size_t)bidx * 64 + col) * N + (size_t)(nb0 + vv)] = (bf16)(cv[vv] + bvc);
    }
  }
}

// ---------------- K2: q = LN(slots)@Wq^T + bq  (bf16, rows 7..15 zero-padded) ----------------
__global__ __launch_bounds__(256)
void k_qproj(const float* __restrict__ slots, const float* __restrict__ Wq,
             const float* __restrict__ bq, const float* __restrict__ gsl,
             const float* __restrict__ bsl, bf16* __restrict__ qb) {
  __shared__ float sn[224 * 64];
  const int t = threadIdx.x;
  for (int i = t; i < B * 16 * 64; i += 256)
    if (((i >> 6) & 15) >= NS) qb[i] = (bf16)0.f;
  if (t < 224) {
    const float* r = slots + (size_t)t * 64;
    float m = 0.f;
    for (int d = 0; d < 64; ++d) m += r[d];
    m *= (1.f / 64.f);
    float v = 0.f;
    for (int d = 0; d < 64; ++d) { float dd = r[d] - m; v += dd * dd; }
    v *= (1.f / 64.f);
    float rs = rsqrtf(v + 1e-5f);
    for (int d = 0; d < 64; ++d) sn[t * 64 + d] = (r[d] - m) * rs * gsl[d] + bsl[d];
  }
  __syncthreads();
  if (t < 224) {
    int b = t / NS, s = t % NS;
    for (int nn = 0; nn < 64; ++nn) {
      float acc = bq[nn];
      for (int d = 0; d < 64; ++d) acc += sn[t * 64 + d] * Wq[nn * 64 + d];
      qb[((size_t)(b * 16 + s)) * 64 + nn] = (bf16)acc;
    }
  }
}

// ---------------- K3: dots = q@k^T * SCALE, softmax over slots, attn bf16 + partial rowsums ----------------
__global__ __launch_bounds__(256)
void k_dots(const bf16* __restrict__ qb, const bf16* __restrict__ kbuf,
            bf16* __restrict__ attn, float* __restrict__ partial) {
  const int b = blockIdx.x >> 7;
  const int chunk = blockIdx.x & 127;
  const int w = threadIdx.x >> 5;
  const int lane = threadIdx.x & 31;
  const int key0 = chunk * 128 + w * 16;

  const bf16* qbase = qb + (size_t)b * 16 * 64;
  v16bf a0 = ldfrag(qbase, 64);
  v16bf a1 = ldfrag(qbase + 32, 64);
  const bf16* kb = kbuf + ((size_t)b * N + key0) * 64;
  v16bf b0 = ldfrag(kb, 64);
  v16bf b1 = ldfrag(kb + 32, 64);
  v8f c = {};
  c = WMMA_BF16(a0, b0, c);
  c = WMMA_BF16(a1, b1, c);

  float sv[NS];
#pragma unroll
  for (int i = 0; i < NS; ++i) sv[i] = 0.f;
  const int j = key0 + (lane & 15);
  bf16* acol = attn + (size_t)b * 16 * N + j;
  if (lane < 16) {
    float mx = c[0] * SCALE;
#pragma unroll
    for (int i = 1; i < NS; ++i) mx = fmaxf(mx, c[i] * SCALE);
    float e[NS]; float s = 0.f;
#pragma unroll
    for (int i = 0; i < NS; ++i) { e[i] = __expf(c[i] * SCALE - mx); s += e[i]; }
    float inv = 1.f / s;
#pragma unroll
    for (int i = 0; i < NS; ++i) {
      bf16 av = (bf16)(e[i] * inv + EPS);
      acol[(size_t)i * N] = av;
      sv[i] = (float)av;
    }
    acol[(size_t)7 * N] = (bf16)0.f;
  } else {
#pragma unroll
    for (int i = 8; i < 16; ++i) acol[(size_t)i * N] = (bf16)0.f;
  }
#pragma unroll
  for (int i = 0; i < NS; ++i) {
    float r = sv[i];
    r += __shfl_xor(r, 1, 32);
    r += __shfl_xor(r, 2, 32);
    r += __shfl_xor(r, 4, 32);
    r += __shfl_xor(r, 8, 32);
    if (lane == 0) partial[((size_t)(b * NS + i)) * 1024 + chunk * 8 + w] = r;
  }
}

// ---------------- K4: rowsum over 1024 per-wave partials (deterministic) ----------------
__global__ void k_rowsum(const float* __restrict__ partial, float* __restrict__ rowsum) {
  int t = blockIdx.x * 256 + threadIdx.x;
  if (t < B * NS) {
    const float* p = partial + (size_t)t * 1024;
    float s = 0.f;
    for (int i = 0; i < 1024; ++i) s += p[i];
    rowsum[t] = s;
  }
}

// ---------------- K5: updates = (attn @ v) / rowsum  (K=16384 split over 8 waves) ----------------
__global__ __launch_bounds__(256)
void k_updates(const bf16* __restrict__ attn, const bf16* __restrict__ vT,
               const float* __restrict__ rowsum, float* __restrict__ upd) {
  __shared__ float red[8 * 256];
  const int b = blockIdx.x >> 2;
  const int dt = blockIdx.x & 3;
  const int w = threadIdx.x >> 5;
  const int lane = threadIdx.x & 31;
  const bf16* ab = attn + (size_t)b * 16 * N + w * 2048;
  const bf16* vb = vT + ((size_t)b * 64 + dt * 16) * N + w * 2048;
  v8f c = {};
  for (int ks = 0; ks < 64; ++ks) {
    v16bf af = ldfrag(ab + ks * 32, N);
    v16bf bf_ = ldfrag(vb + ks * 32, N);
    c = WMMA_BF16(af, bf_, c);
  }
  const int n = lane & 15;
  const int mb = (lane >> 4) << 3;
#pragma unroll
  for (int vv = 0; vv < 8; ++vv) red[w * 256 + (mb + vv) * 16 + n] = c[vv];
  __syncthreads();
  const int m = threadIdx.x >> 4;
  const int nn = threadIdx.x & 15;
  float s = 0.f;
#pragma unroll
  for (int ww = 0; ww < 8; ++ww) s += red[ww * 256 + threadIdx.x];
  if (m < NS) upd[((size_t)b * 16 + m) * 64 + dt * 16 + nn] = s / rowsum[b * NS + m];
}

// ---------------- K6a: GRUCell -> hnew ----------------
__global__ __launch_bounds__(64)
void k_gru(const float* __restrict__ upd, const float* __restrict__ slots,
           const float* __restrict__ Wih, const float* __restrict__ Whh,
           const float* __restrict__ bih, const float* __restrict__ bhh,
           float* __restrict__ hnew) {
  __shared__ float u[64], h[64];
  const int row = blockIdx.x;            // 0..223
  const int bidx = row / NS, s = row % NS;
  const int k = threadIdx.x;
  u[k] = upd[((size_t)bidx * 16 + s) * 64 + k];
  h[k] = slots[(size_t)row * 64 + k];
  __syncthreads();
  float ir = bih[k], hr = bhh[k], iz = bih[64 + k], hz = bhh[64 + k],
        in_ = bih[128 + k], hn = bhh[128 + k];
  for (int d = 0; d < 64; ++d) {
    float ud = u[d], hd = h[d];
    ir  += ud * Wih[k * 64 + d];          hr += hd * Whh[k * 64 + d];
    iz  += ud * Wih[(64 + k) * 64 + d];   hz += hd * Whh[(64 + k) * 64 + d];
    in_ += ud * Wih[(128 + k) * 64 + d];  hn += hd * Whh[(128 + k) * 64 + d];
  }
  float r = 1.f / (1.f + __expf(-(ir + hr)));
  float z = 1.f / (1.f + __expf(-(iz + hz)));
  float nv = tanhf(in_ + r * hn);
  hnew[(size_t)row * 64 + k] = (1.f - z) * nv + z * h[k];
}

// ---------------- K6b: slots = hnew + MLP(LN(hnew)) ----------------
__global__ __launch_bounds__(64)
void k_mlp(const float* __restrict__ hnew, const float* __restrict__ mW1,
           const float* __restrict__ mb1, const float* __restrict__ mW2,
           const float* __restrict__ mb2, const float* __restrict__ gml,
           const float* __restrict__ bml, float* __restrict__ slots) {
  __shared__ float hv[64], mv[64], hid[128];
  const int row = blockIdx.x;
  const int d = threadIdx.x;
  hv[d] = hnew[(size_t)row * 64 + d];
  __syncthreads();
  float m = 0.f;
  for (int i = 0; i < 64; ++i) m += hv[i];
  m *= (1.f / 64.f);
  float v = 0.f;
  for (int i = 0; i < 64; ++i) { float dd = hv[i] - m; v += dd * dd; }
  v *= (1.f / 64.f);
  float rs = rsqrtf(v + 1e-5f);
  mv[d] = (hv[d] - m) * rs * gml[d] + bml[d];
  __syncthreads();
  for (int jj = d; jj < 128; jj += 64) {
    float acc = mb1[jj];
    for (int x = 0; x < 64; ++x) acc += mv[x] * mW1[jj * 64 + x];
    hid[jj] = fmaxf(acc, 0.f);
  }
  __syncthreads();
  float o = mb2[d];
  for (int hh = 0; hh < 128; ++hh) o += hid[hh] * mW2[d * 128 + hh];
  slots[(size_t)row * 64 + d] = hv[d] + o;
}

extern "C" void kernel_launch(void* const* d_in, const int* in_sizes, int n_in,
                              void* d_out, int out_size, void* d_ws, size_t ws_size,
                              hipStream_t stream) {
  const float* inputs = (const float*)d_in[0];
  const float* noise  = (const float*)d_in[1];
  const float* mu     = (const float*)d_in[2];
  const float* sg     = (const float*)d_in[3];
  const float* Wq     = (const float*)d_in[4];
  const float* bq     = (const float*)d_in[5];
  const float* Wk     = (const float*)d_in[6];
  const float* bk     = (const float*)d_in[7];
  const float* Wv     = (const float*)d_in[8];
  const float* bv     = (const float*)d_in[9];
  const float* Wih    = (const float*)d_in[10];
  const float* Whh    = (const float*)d_in[11];
  const float* bih    = (const float*)d_in[12];
  const float* bhh    = (const float*)d_in[13];
  const float* mW1    = (const float*)d_in[14];
  const float* mb1    = (const float*)d_in[15];
  const float* mW2    = (const float*)d_in[16];
  const float* mb2    = (const float*)d_in[17];
  const float* gin    = (const float*)d_in[18];
  const float* bin    = (const float*)d_in[19];
  const float* gsl    = (const float*)d_in[20];
  const float* bsl    = (const float*)d_in[21];
  const float* gml    = (const float*)d_in[22];
  const float* bml    = (const float*)d_in[23];

  char* ws = (char*)d_ws;
  bf16* kbuf = (bf16*)ws;  ws += (size_t)B * N * D * sizeof(bf16);     // 64 MB
  bf16* vT   = (bf16*)ws;  ws += (size_t)B * D * N * sizeof(bf16);     // 64 MB
  bf16* attn = (bf16*)ws;  ws += (size_t)B * 16 * N * sizeof(bf16);    // 16 MB
  bf16* qb   = (bf16*)ws;  ws += (size_t)B * 16 * 64 * sizeof(bf16);
  float* partial = (float*)ws; ws += (size_t)B * NS * 1024 * sizeof(float);
  float* rowsum  = (float*)ws; ws += (size_t)B * NS * sizeof(float);
  float* upd     = (float*)ws; ws += (size_t)B * 16 * 64 * sizeof(float);
  float* hnewb   = (float*)ws;

  float* slots = (float*)d_out;

  k_init_slots<<<(B * NS * D + 255) / 256, 256, 0, stream>>>(noise, mu, sg, slots);
  k_ln_kv<<<(B * N) / 128, 256, 0, stream>>>(inputs, Wk, bk, Wv, bv, gin, bin, kbuf, vT);
  for (int it = 0; it < 3; ++it) {
    k_qproj<<<1, 256, 0, stream>>>(slots, Wq, bq, gsl, bsl, qb);
    k_dots<<<B * 128, 256, 0, stream>>>(qb, kbuf, attn, partial);
    k_rowsum<<<1, 256, 0, stream>>>(partial, rowsum);
    k_updates<<<B * 4, 256, 0, stream>>>(attn, vT, rowsum, upd);
    k_gru<<<B * NS, 64, 0, stream>>>(upd, slots, Wih, Whh, bih, bhh, hnewb);
    k_mlp<<<B * NS, 64, 0, stream>>>(hnewb, mW1, mb1, mW2, mb2, gml, bml, slots);
  }
}
